// AtlasMTLModel_83975200571720
// MI455X (gfx1250) — matrix operand order repaired and checked by reading, not compiled, not run on hardware
//
#include <hip/hip_runtime.h>
#include <hip/hip_bf16.h>

// ---------------- problem constants ----------------
#define BATCH   65536
#define INDIM   2000
#define INPAD   2016      // 63 * 32
#define H1DIM   512
#define H2DIM   256
#define PDIM    30
#define CDIM    200
#define NHOT    4
#define CHDIM   12
#define FHDIM   64
#define FEATDIM 298       // H2 + P + CH
#define FEATPAD 320       // 10 * 32
#define NHEADS  248       // P + C + 16 + 2
#define NHEADP  256

typedef __bf16 bf16;
typedef __attribute__((ext_vector_type(4)))  __bf16 bf16x4;
typedef __attribute__((ext_vector_type(8)))  __bf16 bf16x8;
typedef __attribute__((ext_vector_type(16))) __bf16 bf16x16;
typedef __attribute__((ext_vector_type(4)))  float  f32x4;
typedef __attribute__((ext_vector_type(8)))  float  f32x8;

// ---------------- WMMA fragment helpers ----------------
// A fragment (16x32 bf16, row-major source):
//   lane L holds row M = L&15.  lanes 0-15: K 0..15;
//   lanes 16-31: K 16..31.  (contiguous order, matches load_fragB)
__device__ __forceinline__ bf16x16 load_fragA(const bf16* prow_k0, int hi) {
  bf16x8 c0 = *(const bf16x8*)(prow_k0 + hi * 16);
  bf16x8 c1 = *(const bf16x8*)(prow_k0 + hi * 16 + 8);
  bf16x16 r;
#pragma unroll
  for (int i = 0; i < 8; ++i) { r[i] = c0[i]; r[i + 8] = c1[i]; }
  return r;
}

// B fragment (32x16 bf16) from K-major (transposed) weights Wt[n][k]:
//   lane L holds column N = L&15; lanes 0-15 -> K 0..15, lanes 16-31 -> K 16..31.
__device__ __forceinline__ bf16x16 load_fragB(const bf16* pcol_k) {
  bf16x8 c0 = *(const bf16x8*)(pcol_k);
  bf16x8 c1 = *(const bf16x8*)(pcol_k + 8);
  bf16x16 r;
#pragma unroll
  for (int i = 0; i < 8; ++i) { r[i] = c0[i]; r[i + 8] = c1[i]; }
  return r;
}

__device__ __forceinline__ f32x8 wmma_bf16(bf16x16 a, bf16x16 b, f32x8 c) {
  return __builtin_amdgcn_wmma_f32_16x16x32_bf16(false, a, false, b, (short)0, c,
                                                 false, false);
}

__device__ __forceinline__ bf16x4 cvt4(f32x4 v) {
  bf16x4 o;
  o[0] = (bf16)v[0]; o[1] = (bf16)v[1]; o[2] = (bf16)v[2]; o[3] = (bf16)v[3];
  return o;
}

// ---------------- weight prep (fp32 -> padded transposed bf16) ----------------
__global__ void k_prep_wt1(const float* __restrict__ W1, bf16* __restrict__ wt1) {
  int i = blockIdx.x * blockDim.x + threadIdx.x;
  if (i >= H1DIM * INPAD) return;
  int n = i / INPAD, k = i % INPAD;
  wt1[i] = (k < INDIM) ? (bf16)W1[(size_t)k * H1DIM + n] : (bf16)0.f;
}

__global__ void k_prep_wt2(const float* __restrict__ W2, bf16* __restrict__ wt2) {
  int i = blockIdx.x * blockDim.x + threadIdx.x;
  if (i >= H2DIM * H1DIM) return;
  int n = i / H1DIM, k = i % H1DIM;
  wt2[i] = (bf16)W2[(size_t)k * H2DIM + n];
}

__global__ void k_prep_heads(const float* __restrict__ Wp, const float* __restrict__ Wc,
                             const float* __restrict__ Wlat, const float* __restrict__ Wum,
                             const float* __restrict__ bp, const float* __restrict__ bc,
                             const float* __restrict__ blat, const float* __restrict__ bum,
                             bf16* __restrict__ wh, float* __restrict__ bh) {
  int i = blockIdx.x * blockDim.x + threadIdx.x;
  if (i >= NHEADP * H2DIM) return;
  int n = i / H2DIM, k = i % H2DIM;
  float v;
  if (n < PDIM)                  v = Wp[(size_t)k * PDIM + n];
  else if (n < PDIM + CDIM)      v = Wc[(size_t)k * CDIM + (n - PDIM)];
  else if (n < PDIM + CDIM + 16) v = Wlat[(size_t)k * 16 + (n - PDIM - CDIM)];
  else if (n < NHEADS)           v = Wum[(size_t)k * 2 + (n - PDIM - CDIM - 16)];
  else                           v = 0.f;
  wh[i] = (bf16)v;
  if (k == 0) {
    float b;
    if (n < PDIM)                  b = bp[n];
    else if (n < PDIM + CDIM)      b = bc[n - PDIM];
    else if (n < PDIM + CDIM + 16) b = blat[n - PDIM - CDIM];
    else if (n < NHEADS)           b = bum[n - PDIM - CDIM - 16];
    else                           b = 0.f;
    bh[n] = b;
  }
}

__global__ void k_prep_expert(const float* __restrict__ cW1, const float* __restrict__ cW2,
                              bf16* __restrict__ cw1t, bf16* __restrict__ cw2t) {
  int i = blockIdx.x * blockDim.x + threadIdx.x;
  const int n1 = NHOT * FHDIM * FEATPAD;       // cw1t elements
  const int n2 = NHOT * 16 * FHDIM;            // cw2t elements
  if (i < n1) {
    int f = i % FEATPAD;
    int h = (i / FEATPAD) % FHDIM;
    int n = i / (FEATPAD * FHDIM);
    cw1t[i] = (f < FEATDIM) ? (bf16)cW1[((size_t)n * FEATDIM + f) * FHDIM + h]
                            : (bf16)0.f;
  } else if (i < n1 + n2) {
    int j = i - n1;
    int h = j % FHDIM;
    int c = (j / FHDIM) % 16;
    int n = j / (FHDIM * 16);
    cw2t[j] = (c < CHDIM) ? (bf16)cW2[((size_t)n * FHDIM + h) * CHDIM + c]
                          : (bf16)0.f;
  }
}

// ---------------- GEMM1: z1 = relu(x @ W1 + b1) ----------------
// block = 256 threads (8 waves), tile = 64 rows x 512 cols.
// wave w: all 4 M-tiles x 4 N-tiles (cols w*64 .. w*64+63); acc = 128 VGPRs.
// Each B fragment from L2 is reused by 4 WMMAs (M-tiles) -> 4x less W1 traffic.
__global__ __launch_bounds__(256) void k_gemm1(const float* __restrict__ x,
                                               const bf16* __restrict__ wt1,
                                               const float* __restrict__ b1,
                                               bf16* __restrict__ z1) {
  __shared__ __attribute__((aligned(16))) bf16 As[64 * 32];
  const int tid = threadIdx.x, lane = tid & 31, wave = tid >> 5;
  const int hi = lane >> 4;
  const int row0 = blockIdx.x * 64;
  f32x8 acc[4][4] = {};
  for (int kt = 0; kt < INPAD / 32; ++kt) {
    const int k0 = kt * 32;
    __syncthreads();
    // stage 64x32 fp32 -> bf16 via float4 loads + packed converts + ds_store_b64.
    // chunks of 4 never straddle the 2000-element boundary (2000 % 4 == 0).
#pragma unroll
    for (int c = tid; c < 64 * 8; c += 256) {
      int r = c >> 3, kc = (c & 7) * 4;
      int gk = k0 + kc;
      f32x4 v = {0.f, 0.f, 0.f, 0.f};
      if (gk < INDIM) v = *(const f32x4*)&x[(size_t)(row0 + r) * INDIM + gk];
      *(bf16x4*)&As[r * 32 + kc] = cvt4(v);
    }
    // prefetch next K slab of x (one cacheline per row) -> global_prefetch_b8
    if (tid < 64 && k0 + 32 < INDIM)
      __builtin_prefetch(&x[(size_t)(row0 + tid) * INDIM + k0 + 32], 0, 1);
    __syncthreads();
    bf16x16 a[4];
#pragma unroll
    for (int mt = 0; mt < 4; ++mt)
      a[mt] = load_fragA(&As[(mt * 16 + (lane & 15)) * 32], hi);
#pragma unroll
    for (int nt = 0; nt < 4; ++nt) {
      int n = wave * 64 + nt * 16 + (lane & 15);
      bf16x16 b = load_fragB(wt1 + (size_t)n * INPAD + k0 + hi * 16);
#pragma unroll
      for (int mt = 0; mt < 4; ++mt)
        acc[mt][nt] = wmma_bf16(a[mt], b, acc[mt][nt]);
    }
  }
#pragma unroll
  for (int nt = 0; nt < 4; ++nt) {
    int n = wave * 64 + nt * 16 + (lane & 15);
    float bias = b1[n];
#pragma unroll
    for (int mt = 0; mt < 4; ++mt) {
#pragma unroll
      for (int r = 0; r < 8; ++r) {
        int row = row0 + mt * 16 + r + hi * 8;
        float v = acc[mt][nt][r] + bias;
        v = v > 0.f ? v : 0.f;
        z1[(size_t)row * H1DIM + n] = (bf16)v;
      }
    }
  }
}

// ---------------- GEMM2: z = relu(z1 @ W2 + b2) -> fp32 out + bf16 ws ----------------
// block = 256 threads (8 waves), tile = 64 rows x 256 cols; wave = 4 M x 2 N tiles.
__global__ __launch_bounds__(256) void k_gemm2(const bf16* __restrict__ z1,
                                               const bf16* __restrict__ wt2,
                                               const float* __restrict__ b2,
                                               float* __restrict__ z_out,
                                               bf16* __restrict__ z_bf) {
  const int tid = threadIdx.x, lane = tid & 31, wave = tid >> 5;
  const int hi = lane >> 4;
  const int row0 = blockIdx.x * 64;
  f32x8 acc[4][2] = {};
  for (int kt = 0; kt < H1DIM / 32; ++kt) {
    const int k0 = kt * 32;
    bf16x16 a[4];
#pragma unroll
    for (int mt = 0; mt < 4; ++mt)
      a[mt] = load_fragA(z1 + (size_t)(row0 + mt * 16 + (lane & 15)) * H1DIM + k0, hi);
#pragma unroll
    for (int nt = 0; nt < 2; ++nt) {
      int n = wave * 32 + nt * 16 + (lane & 15);
      bf16x16 b = load_fragB(wt2 + (size_t)n * H1DIM + k0 + hi * 16);
#pragma unroll
      for (int mt = 0; mt < 4; ++mt)
        acc[mt][nt] = wmma_bf16(a[mt], b, acc[mt][nt]);
    }
  }
#pragma unroll
  for (int nt = 0; nt < 2; ++nt) {
    int n = wave * 32 + nt * 16 + (lane & 15);
    float bias = b2[n];
#pragma unroll
    for (int mt = 0; mt < 4; ++mt) {
#pragma unroll
      for (int r = 0; r < 8; ++r) {
        int row = row0 + mt * 16 + r + hi * 8;
        float v = acc[mt][nt][r] + bias;
        v = v > 0.f ? v : 0.f;
        z_out[(size_t)row * H2DIM + n] = v;
        z_bf[(size_t)row * H2DIM + n] = (bf16)v;
      }
    }
  }
}

// ---------------- Heads GEMM: packed [Wp|Wc|Wlat|Wum], routed epilogue ----------------
__global__ __launch_bounds__(256) void k_heads(const bf16* __restrict__ zb,
                                               const bf16* __restrict__ wh,
                                               const float* __restrict__ bh,
                                               float* __restrict__ out_parent,
                                               float* __restrict__ out_child,
                                               float* __restrict__ child_ws,
                                               float* __restrict__ out_lat,
                                               float* __restrict__ out_um) {
  const int tid = threadIdx.x, lane = tid & 31, wave = tid >> 5;
  const int hi = lane >> 4;
  const int row0 = blockIdx.x * 64;
  f32x8 acc[4][2] = {};
  for (int kt = 0; kt < H2DIM / 32; ++kt) {
    const int k0 = kt * 32;
    bf16x16 a[4];
#pragma unroll
    for (int mt = 0; mt < 4; ++mt)
      a[mt] = load_fragA(zb + (size_t)(row0 + mt * 16 + (lane & 15)) * H2DIM + k0, hi);
#pragma unroll
    for (int nt = 0; nt < 2; ++nt) {
      int n = wave * 32 + nt * 16 + (lane & 15);
      bf16x16 b = load_fragB(wh + (size_t)n * H2DIM + k0 + hi * 16);
#pragma unroll
      for (int mt = 0; mt < 4; ++mt)
        acc[mt][nt] = wmma_bf16(a[mt], b, acc[mt][nt]);
    }
  }
#pragma unroll
  for (int nt = 0; nt < 2; ++nt) {
    int n = wave * 32 + nt * 16 + (lane & 15);
    float bias = bh[n];
#pragma unroll
    for (int mt = 0; mt < 4; ++mt) {
#pragma unroll
      for (int r = 0; r < 8; ++r) {
        int row = row0 + mt * 16 + r + hi * 8;
        float v = acc[mt][nt][r] + bias;
        if (n < PDIM) {
          out_parent[(size_t)row * PDIM + n] = v;
        } else if (n < PDIM + CDIM) {
          int c = n - PDIM;
          out_child[(size_t)row * CDIM + c] = v;
          child_ws[(size_t)row * CDIM + c] = v;
        } else if (n < PDIM + CDIM + 16) {
          out_lat[(size_t)row * 16 + (n - PDIM - CDIM)] = v;
        } else if (n < NHEADS) {
          out_um[(size_t)row * 2 + (n - PDIM - CDIM - 16)] = v;
        }
      }
    }
  }
}

// ---------------- argmax + hotspot mask ----------------
__global__ void k_argmax(const float* __restrict__ parent, const int* __restrict__ hotspot_ids,
                         int* __restrict__ pidx, float* __restrict__ out_mask) {
  int row = blockIdx.x * blockDim.x + threadIdx.x;
  if (row >= BATCH) return;
  const float* p = parent + (size_t)row * PDIM;
  float best = p[0];
  int bi = 0;
#pragma unroll
  for (int i = 1; i < PDIM; ++i) {
    float v = p[i];
    if (v > best) { best = v; bi = i; }
  }
  pidx[row] = bi;
  float act = 0.f;
#pragma unroll
  for (int n = 0; n < NHOT; ++n)
    if (hotspot_ids[n] == bi) act = 1.f;
  out_mask[row] = act;
}

// ---------------- hotspot expert MLP (all-WMMA) ----------------
// 128 threads = 4 waves, wave n = hotspot n; tile = 16 rows.
__global__ __launch_bounds__(128) void k_hotspot(
    const float* __restrict__ z, const float* __restrict__ parent,
    const float* __restrict__ child_ws, const bf16* __restrict__ cw1t,
    const bf16* __restrict__ cw2t, const float* __restrict__ cb1,
    const float* __restrict__ cb2, const int* __restrict__ child_idx,
    const int* __restrict__ hotspot_ids, const int* __restrict__ pidx,
    float* __restrict__ out_child) {
  __shared__ __attribute__((aligned(16))) bf16 F[NHOT][16 * FEATPAD];
  __shared__ __attribute__((aligned(16))) bf16 Ht[NHOT][16 * FHDIM];
  const int tid = threadIdx.x, lane = tid & 31, n = tid >> 5;
  const int hi = lane >> 4;
  const int row0 = blockIdx.x * 16;

  // feats = [z(256) | parent(30) | child subset(12) | zero pad] as bf16.
  // z portion vectorized: float4 load + packed cvt + ds_store_b64 per chunk.
  for (int c = lane; c < 16 * (H2DIM / 4); c += 32) {
    int r = c >> 6, fc = (c & 63) * 4;
    f32x4 v = *(const f32x4*)&z[(size_t)(row0 + r) * H2DIM + fc];
    *(bf16x4*)&F[n][r * FEATPAD + fc] = cvt4(v);
  }
  // tail: parent logits, child subset, pad (64 columns per row)
  for (int e = lane; e < 16 * 64; e += 32) {
    int r = e >> 6, f = H2DIM + (e & 63);
    int row = row0 + r;
    float v;
    if (f < H2DIM + PDIM) v = parent[(size_t)row * PDIM + (f - H2DIM)];
    else if (f < FEATDIM) v = child_ws[(size_t)row * CDIM +
                                       child_idx[n * CHDIM + (f - H2DIM - PDIM)]];
    else                  v = 0.f;
    F[n][r * FEATPAD + f] = (bf16)v;
  }
  __syncthreads();

  // layer 1: h = relu(feats @ cW1[n] + cb1[n])   (16x320 @ 320x64)
  f32x8 acc1[4] = {};
  for (int kt = 0; kt < FEATPAD / 32; ++kt) {
    const int k0 = kt * 32;
    bf16x16 a = load_fragA(&F[n][(lane & 15) * FEATPAD + k0], hi);
#pragma unroll
    for (int t = 0; t < 4; ++t) {
      int h = t * 16 + (lane & 15);
      bf16x16 b = load_fragB(cw1t + ((size_t)(n * FHDIM + h)) * FEATPAD + k0 + hi * 16);
      acc1[t] = wmma_bf16(a, b, acc1[t]);
    }
  }
#pragma unroll
  for (int t = 0; t < 4; ++t) {
    int h = t * 16 + (lane & 15);
    float bias = cb1[n * FHDIM + h];
#pragma unroll
    for (int r = 0; r < 8; ++r) {
      int hrow = r + hi * 8;
      float v = acc1[t][r] + bias;
      v = v > 0.f ? v : 0.f;
      Ht[n][hrow * FHDIM + h] = (bf16)v;
    }
  }
  __syncthreads();

  // layer 2: residual = h @ cW2[n] + cb2[n]      (16x64 @ 64x16, 12 valid cols)
  f32x8 acc2 = {};
  for (int kt = 0; kt < FHDIM / 32; ++kt) {
    const int k0 = kt * 32;
    bf16x16 a = load_fragA(&Ht[n][(lane & 15) * FHDIM + k0], hi);
    bf16x16 b = load_fragB(cw2t + ((size_t)(n * 16 + (lane & 15))) * FHDIM + k0 + hi * 16);
    acc2 = wmma_bf16(a, b, acc2);
  }

  const int c = lane & 15;
  if (c < CHDIM) {
    int cidx = child_idx[n * CHDIM + c];
    float bias = cb2[n * CHDIM + c];
    int hid = hotspot_ids[n];
#pragma unroll
    for (int r = 0; r < 8; ++r) {
      int row = row0 + r + hi * 8;
      float m = (pidx[row] == hid) ? 1.f : 0.f;
      float v = child_ws[(size_t)row * CDIM + cidx] + (acc2[r] + bias) * 1.0f * m;
      out_child[(size_t)row * CDIM + cidx] = v;
    }
  }
}

// ---------------- host launcher ----------------
extern "C" void kernel_launch(void* const* d_in, const int* in_sizes, int n_in,
                              void* d_out, int out_size, void* d_ws, size_t ws_size,
                              hipStream_t stream) {
  const float* x    = (const float*)d_in[0];
  const float* W1   = (const float*)d_in[1];
  const float* b1   = (const float*)d_in[2];
  const float* W2   = (const float*)d_in[3];
  const float* b2   = (const float*)d_in[4];
  const float* Wp   = (const float*)d_in[5];
  const float* bp   = (const float*)d_in[6];
  const float* Wc   = (const float*)d_in[7];
  const float* bc   = (const float*)d_in[8];
  const float* Wlat = (const float*)d_in[9];
  const float* blat = (const float*)d_in[10];
  const float* Wum  = (const float*)d_in[11];
  const float* bum  = (const float*)d_in[12];
  const float* cW1  = (const float*)d_in[13];
  const float* cb1  = (const float*)d_in[14];
  const float* cW2  = (const float*)d_in[15];
  const float* cb2  = (const float*)d_in[16];
  const int* hotspot_ids = (const int*)d_in[17];
  const int* child_idx   = (const int*)d_in[18];

  // d_out regions (flat concatenation in return order)
  float* out = (float*)d_out;
  float* out_parent = out;
  float* out_child  = out_parent + (size_t)BATCH * PDIM;
  float* out_lat    = out_child + (size_t)BATCH * CDIM;
  float* out_um     = out_lat + (size_t)BATCH * 16;
  float* out_z      = out_um + (size_t)BATCH * 2;
  float* out_mask   = out_z + (size_t)BATCH * H2DIM;

  // workspace carve-up (256B aligned)
  char* w = (char*)d_ws;
  size_t o = 0;
  auto carve = [&](size_t bytes) -> char* {
    char* p = w + o;
    o += (bytes + 255) & ~(size_t)255;
    return p;
  };
  bf16*  wt1      = (bf16*)carve((size_t)H1DIM * INPAD * 2);
  bf16*  wt2      = (bf16*)carve((size_t)H2DIM * H1DIM * 2);
  bf16*  wh       = (bf16*)carve((size_t)NHEADP * H2DIM * 2);
  float* bh       = (float*)carve((size_t)NHEADP * 4);
  bf16*  cw1t     = (bf16*)carve((size_t)NHOT * FHDIM * FEATPAD * 2);
  bf16*  cw2t     = (bf16*)carve((size_t)NHOT * 16 * FHDIM * 2);
  bf16*  z1       = (bf16*)carve((size_t)BATCH * H1DIM * 2);
  bf16*  zbf      = (bf16*)carve((size_t)BATCH * H2DIM * 2);
  float* child_ws = (float*)carve((size_t)BATCH * CDIM * 4);
  int*   pidx     = (int*)carve((size_t)BATCH * 4);

  // weight prep (tiny)
  k_prep_wt1<<<(H1DIM * INPAD + 255) / 256, 256, 0, stream>>>(W1, wt1);
  k_prep_wt2<<<(H2DIM * H1DIM + 255) / 256, 256, 0, stream>>>(W2, wt2);
  k_prep_heads<<<(NHEADP * H2DIM + 255) / 256, 256, 0, stream>>>(
      Wp, Wc, Wlat, Wum, bp, bc, blat, bum, wh, bh);
  {
    int total = NHOT * FHDIM * FEATPAD + NHOT * 16 * FHDIM;
    k_prep_expert<<<(total + 255) / 256, 256, 0, stream>>>(cW1, cW2, cw1t, cw2t);
  }

  k_gemm1<<<BATCH / 64, 256, 0, stream>>>(x, wt1, b1, z1);
  k_gemm2<<<BATCH / 64, 256, 0, stream>>>(z1, wt2, b2, out_z, zbf);
  k_heads<<<BATCH / 64, 256, 0, stream>>>(zbf, wh, bh, out_parent, out_child,
                                          child_ws, out_lat, out_um);
  k_argmax<<<BATCH / 256, 256, 0, stream>>>(out_parent, hotspot_ids, pidx, out_mask);
  k_hotspot<<<BATCH / 16, 128, 0, stream>>>(out_z, out_parent, child_ws, cw1t, cw2t,
                                            cb1, cb2, child_idx, hotspot_ids, pidx,
                                            out_child);
}
